// MultiHeadAttention_31293131719321
// MI455X (gfx1250) — matrix operand order repaired and checked by reading, not compiled, not run on hardware
//
#include <hip/hip_runtime.h>
#include <hip/hip_bf16.h>

typedef __attribute__((ext_vector_type(16))) __bf16 v16bf;
typedef __attribute__((ext_vector_type(8)))  __bf16 v8bf;
typedef __attribute__((ext_vector_type(8)))  float  v8f;

#define BB 2
#define SS 2048
#define DDIM 1024
#define HH 16
#define DH 64
#define NEG_INF -1e20f

static __device__ inline v16bf cat8(v8bf lo, v8bf hi) {
  return __builtin_shufflevector(lo, hi, 0,1,2,3,4,5,6,7,8,9,10,11,12,13,14,15);
}

static __device__ inline v8f wmma_bf16(v16bf a, v16bf b, v8f c) {
  return __builtin_amdgcn_wmma_f32_16x16x32_bf16(false, a, false, b, (short)0, c, false, false);
}

// A-fragment (16x32 bf16) from a row-major row pointer, k-offset kb.
static __device__ inline v16bf load_afrag(const __bf16* row, int kb, int half) {
  v8bf a0 = *(const v8bf*)(row + kb + half * 16);
  v8bf a1 = *(const v8bf*)(row + kb + half * 16 + 8);
  return cat8(a0, a1);
}

// ---------------- fp32 -> bf16 convert ----------------
__global__ void cvt_bf16(const float* __restrict__ src, __bf16* __restrict__ dst, int n) {
  int i = blockIdx.x * blockDim.x + threadIdx.x;
  if (i < n) dst[i] = (__bf16)src[i];
}

// ---------------- fused QKV projection, 32x64 register-blocked per wave ----------------
// out = x @ W.T + b ; Q scaled by 1/sqrt(dh); Q,K -> [B,H,S,dh]; V -> [B,H,dh,S] (transposed)
__global__ __launch_bounds__(256)
void qkv_proj(const __bf16* __restrict__ xbf,
              const __bf16* __restrict__ wq, const __bf16* __restrict__ wk, const __bf16* __restrict__ wv,
              const float* __restrict__ bq, const float* __restrict__ bk, const float* __restrict__ bv,
              __bf16* __restrict__ Qo, __bf16* __restrict__ Ko, __bf16* __restrict__ Vto) {
  const int lane = threadIdx.x & 31, wave = threadIdx.x >> 5;
  const int half = lane >> 4, mr = lane & 15;
  const int z  = blockIdx.z;
  const int m0 = blockIdx.x * 128 + (wave & 3) * 32;
  const int n0 = blockIdx.y * 128 + (wave >> 2) * 64;

  const __bf16* w    = (z == 0) ? wq : ((z == 1) ? wk : wv);
  const float*  bias = (z == 0) ? bq : ((z == 1) ? bk : bv);

  const __bf16* arow0 = xbf + (size_t)(m0 + mr) * DDIM;
  const __bf16* arow1 = xbf + (size_t)(m0 + 16 + mr) * DDIM;
  const __bf16* brow0 = w + (size_t)(n0 + 0 * 16 + mr) * DDIM;
  const __bf16* brow1 = w + (size_t)(n0 + 1 * 16 + mr) * DDIM;
  const __bf16* brow2 = w + (size_t)(n0 + 2 * 16 + mr) * DDIM;
  const __bf16* brow3 = w + (size_t)(n0 + 3 * 16 + mr) * DDIM;

  v8f acc[2][4] = {};
  for (int kb = 0; kb < DDIM; kb += 32) {
    v16bf A0 = load_afrag(arow0, kb, half);
    v16bf A1 = load_afrag(arow1, kb, half);
    v16bf B0 = *(const v16bf*)(brow0 + kb + half * 16);
    v16bf B1 = *(const v16bf*)(brow1 + kb + half * 16);
    v16bf B2 = *(const v16bf*)(brow2 + kb + half * 16);
    v16bf B3 = *(const v16bf*)(brow3 + kb + half * 16);
    acc[0][0] = wmma_bf16(A0, B0, acc[0][0]);
    acc[0][1] = wmma_bf16(A0, B1, acc[0][1]);
    acc[0][2] = wmma_bf16(A0, B2, acc[0][2]);
    acc[0][3] = wmma_bf16(A0, B3, acc[0][3]);
    acc[1][0] = wmma_bf16(A1, B0, acc[1][0]);
    acc[1][1] = wmma_bf16(A1, B1, acc[1][1]);
    acc[1][2] = wmma_bf16(A1, B2, acc[1][2]);
    acc[1][3] = wmma_bf16(A1, B3, acc[1][3]);
  }

#pragma unroll
  for (int j = 0; j < 4; ++j) {
    const int n = n0 + j * 16 + mr, h = n >> 6, d = n & 63;
    const float bn = bias[n];
#pragma unroll
    for (int i = 0; i < 2; ++i) {
#pragma unroll
      for (int v = 0; v < 8; ++v) {
        int m = m0 + i * 16 + v + 8 * half;
        int bi = m >> 11, s = m & (SS - 1);
        float val = acc[i][j][v] + bn;
        if (z == 0) {
          Qo[((size_t)(bi * HH + h) * SS + s) * DH + d] = (__bf16)(val * 0.125f);
        } else if (z == 1) {
          Ko[((size_t)(bi * HH + h) * SS + s) * DH + d] = (__bf16)val;
        } else {
          Vto[((size_t)(bi * HH + h) * DH + d) * SS + s] = (__bf16)val;
        }
      }
    }
  }
}

// ---------------- attention: single-pass online softmax, per (b,h,16-row q tile) ----------------
__global__ __launch_bounds__(256)
void attn(const __bf16* __restrict__ Qb, const __bf16* __restrict__ Kb,
          const __bf16* __restrict__ Vtb, const unsigned char* __restrict__ mask,
          __bf16* __restrict__ Ao) {
  __shared__ float  red[8][16];     // per-wave running row max
  __shared__ float  sred[8][16];    // per-wave row sum (relative to own max)
  __shared__ float  rowmax[16];
  __shared__ float  rowsum[16];
  __shared__ __bf16 pstage[8][16 * 32];
  __shared__ float  accsm[16 * 64];

  const int tid  = threadIdx.x;
  const int lane = tid & 31, wave = tid >> 5;
  const int half = lane >> 4, mr = lane & 15;
  const int qt = blockIdx.x, h = blockIdx.y, b = blockIdx.z;

  for (int i = tid; i < 16 * 64; i += 256) accsm[i] = 0.f;
  __syncthreads();

  const size_t bh = (size_t)(b * HH + h);
  const __bf16* qrow  = Qb  + (bh * SS + qt * 16 + mr) * DH;
  const __bf16* kbase = Kb  + bh * SS * DH;
  const __bf16* vtb   = Vtb + bh * DH * SS;
  const unsigned char* mrow = mask + (size_t)b * SS;

  // Q A-fragments for the two 32-wide d-chunks
  v16bf aq0 = load_afrag(qrow, 0, half);
  v16bf aq1 = load_afrag(qrow, 32, half);

  float run[8];   // running row max (uniform across each 16-lane half)
  float mysum[8] = {};
#pragma unroll
  for (int v = 0; v < 8; ++v) run[v] = NEG_INF;
  v8f acc0 = {}, acc1 = {}, acc2 = {}, acc3 = {};

  for (int ci = 0; ci < 8; ++ci) {
    const int ch = ci * 8 + wave;   // 64 disjoint chunks of 32 keys
    // ---- scores for the two 16-key tiles of this chunk ----
    v8f sc[2];
#pragma unroll
    for (int t = 0; t < 2; ++t) {
      const int kn = ch * 32 + t * 16 + mr;
      v8f s = {};
      v16bf bm0 = *(const v16bf*)(kbase + (size_t)kn * DH + 0 + half * 16);
      v16bf bm1 = *(const v16bf*)(kbase + (size_t)kn * DH + 32 + half * 16);
      s = wmma_bf16(aq0, bm0, s);
      s = wmma_bf16(aq1, bm1, s);
      const bool mk = mrow[kn] != 0;
#pragma unroll
      for (int v = 0; v < 8; ++v) sc[t][v] = mk ? s[v] : NEG_INF;
    }
    // ---- online max update ----
#pragma unroll
    for (int v = 0; v < 8; ++v) {
      float m = fmaxf(sc[0][v], sc[1][v]);
      m = fmaxf(m, __shfl_xor(m, 1, 16));
      m = fmaxf(m, __shfl_xor(m, 2, 16));
      m = fmaxf(m, __shfl_xor(m, 4, 16));
      m = fmaxf(m, __shfl_xor(m, 8, 16));
      float nm = fmaxf(run[v], m);
      float scale = __expf(run[v] - nm);
      run[v] = nm;
      float p0 = __expf(sc[0][v] - nm);
      float p1 = __expf(sc[1][v] - nm);
      mysum[v] = mysum[v] * scale + p0 + p1;
      acc0[v] *= scale; acc1[v] *= scale; acc2[v] *= scale; acc3[v] *= scale;
      const int r = (v + 8 * half) * 32 + mr;
      pstage[wave][r +  0] = (__bf16)p0;
      pstage[wave][r + 16] = (__bf16)p1;
    }
    __syncthreads();
    // ---- P @ V for this chunk ----
    v8bf p0 = *(const v8bf*)(&pstage[wave][mr * 32 + half * 16]);
    v8bf p1 = *(const v8bf*)(&pstage[wave][mr * 32 + half * 16 + 8]);
    v16bf ap = cat8(p0, p1);
    v16bf bv0 = *(const v16bf*)(vtb + (size_t)(0 * 16 + mr) * SS + ch * 32 + half * 16);
    v16bf bv1 = *(const v16bf*)(vtb + (size_t)(1 * 16 + mr) * SS + ch * 32 + half * 16);
    v16bf bv2 = *(const v16bf*)(vtb + (size_t)(2 * 16 + mr) * SS + ch * 32 + half * 16);
    v16bf bv3 = *(const v16bf*)(vtb + (size_t)(3 * 16 + mr) * SS + ch * 32 + half * 16);
    acc0 = wmma_bf16(ap, bv0, acc0);
    acc1 = wmma_bf16(ap, bv1, acc1);
    acc2 = wmma_bf16(ap, bv2, acc2);
    acc3 = wmma_bf16(ap, bv3, acc3);
    __syncthreads();
  }

  // ---- cross-wave merge: rowwise sums need reduction over 16 lanes first ----
#pragma unroll
  for (int v = 0; v < 8; ++v) {
    float s = mysum[v];
    s += __shfl_xor(s, 1, 16);
    s += __shfl_xor(s, 2, 16);
    s += __shfl_xor(s, 4, 16);
    s += __shfl_xor(s, 8, 16);
    mysum[v] = s;
  }
  if (mr == 0) {
#pragma unroll
    for (int v = 0; v < 8; ++v) {
      red[wave][v + 8 * half]  = run[v];
      sred[wave][v + 8 * half] = mysum[v];
    }
  }
  __syncthreads();
  if (tid < 16) {
    float M = red[0][tid];
#pragma unroll
    for (int w = 1; w < 8; ++w) M = fmaxf(M, red[w][tid]);
    float s = 0.f;
#pragma unroll
    for (int w = 0; w < 8; ++w) s += sred[w][tid] * __expf(red[w][tid] - M);
    rowmax[tid] = M;
    rowsum[tid] = s;
  }
  __syncthreads();

  // scale this wave's attended partials into the global frame and combine (ds_add_f32)
#pragma unroll
  for (int v = 0; v < 8; ++v) {
    const float fac = __expf(run[v] - rowmax[v + 8 * half]);
    const int r = (v + 8 * half) * 64 + mr;
    atomicAdd(&accsm[r + 0 * 16], acc0[v] * fac);
    atomicAdd(&accsm[r + 1 * 16], acc1[v] * fac);
    atomicAdd(&accsm[r + 2 * 16], acc2[v] * fac);
    atomicAdd(&accsm[r + 3 * 16], acc3[v] * fac);
  }
  __syncthreads();

  for (int i = tid; i < 16 * 64; i += 256) {
    int r = i >> 6, c = i & 63;
    float val = accsm[i] / rowsum[r];
    Ao[((size_t)(b * SS + qt * 16 + r)) * DDIM + h * DH + c] = (__bf16)val;
  }
}

// ---------------- output projection: out = attended @ Wo.T + bo (fp32 out), 32x64 blocked ----------------
__global__ __launch_bounds__(256)
void out_proj(const __bf16* __restrict__ abf, const __bf16* __restrict__ wo,
              const float* __restrict__ bo, float* __restrict__ out) {
  const int lane = threadIdx.x & 31, wave = threadIdx.x >> 5;
  const int half = lane >> 4, mr = lane & 15;
  const int m0 = blockIdx.x * 128 + (wave & 3) * 32;
  const int n0 = blockIdx.y * 128 + (wave >> 2) * 64;

  const __bf16* arow0 = abf + (size_t)(m0 + mr) * DDIM;
  const __bf16* arow1 = abf + (size_t)(m0 + 16 + mr) * DDIM;
  const __bf16* brow0 = wo + (size_t)(n0 + 0 * 16 + mr) * DDIM;
  const __bf16* brow1 = wo + (size_t)(n0 + 1 * 16 + mr) * DDIM;
  const __bf16* brow2 = wo + (size_t)(n0 + 2 * 16 + mr) * DDIM;
  const __bf16* brow3 = wo + (size_t)(n0 + 3 * 16 + mr) * DDIM;

  v8f acc[2][4] = {};
  for (int kb = 0; kb < DDIM; kb += 32) {
    v16bf A0 = load_afrag(arow0, kb, half);
    v16bf A1 = load_afrag(arow1, kb, half);
    v16bf B0 = *(const v16bf*)(brow0 + kb + half * 16);
    v16bf B1 = *(const v16bf*)(brow1 + kb + half * 16);
    v16bf B2 = *(const v16bf*)(brow2 + kb + half * 16);
    v16bf B3 = *(const v16bf*)(brow3 + kb + half * 16);
    acc[0][0] = wmma_bf16(A0, B0, acc[0][0]);
    acc[0][1] = wmma_bf16(A0, B1, acc[0][1]);
    acc[0][2] = wmma_bf16(A0, B2, acc[0][2]);
    acc[0][3] = wmma_bf16(A0, B3, acc[0][3]);
    acc[1][0] = wmma_bf16(A1, B0, acc[1][0]);
    acc[1][1] = wmma_bf16(A1, B1, acc[1][1]);
    acc[1][2] = wmma_bf16(A1, B2, acc[1][2]);
    acc[1][3] = wmma_bf16(A1, B3, acc[1][3]);
  }

#pragma unroll
  for (int j = 0; j < 4; ++j) {
    const int n = n0 + j * 16 + mr;
    const float bn = bo[n];
#pragma unroll
    for (int i = 0; i < 2; ++i) {
#pragma unroll
      for (int v = 0; v < 8; ++v) {
        out[(size_t)(m0 + i * 16 + v + 8 * half) * DDIM + n] = acc[i][j][v] + bn;
      }
    }
  }
}

extern "C" void kernel_launch(void* const* d_in, const int* in_sizes, int n_in,
                              void* d_out, int out_size, void* d_ws, size_t ws_size,
                              hipStream_t stream) {
  (void)in_sizes; (void)n_in; (void)out_size; (void)ws_size;
  const float* query = (const float*)d_in[0];
  const unsigned char* mask = (const unsigned char*)d_in[1];
  const float* Wq = (const float*)d_in[2];
  const float* bq = (const float*)d_in[3];
  const float* Wk = (const float*)d_in[4];
  const float* bk = (const float*)d_in[5];
  const float* Wv = (const float*)d_in[6];
  const float* bv = (const float*)d_in[7];
  const float* Wo = (const float*)d_in[8];
  const float* bo = (const float*)d_in[9];

  char* ws = (char*)d_ws;
  size_t off = 0;
  auto alloc = [&](size_t bytes) {
    char* p = ws + off;
    off += (bytes + 255) & ~(size_t)255;
    return p;
  };
  const size_t ACT = (size_t)BB * SS * DDIM;  // 4 Mi elems
  const size_t WEL = (size_t)DDIM * DDIM;     // 1 Mi elems

  __bf16* xbf = (__bf16*)alloc(ACT * 2);
  __bf16* wqb = (__bf16*)alloc(WEL * 2);
  __bf16* wkb = (__bf16*)alloc(WEL * 2);
  __bf16* wvb = (__bf16*)alloc(WEL * 2);
  __bf16* wob = (__bf16*)alloc(WEL * 2);
  __bf16* Qb  = (__bf16*)alloc(ACT * 2);   // [B,H,S,dh], pre-scaled
  __bf16* Kb  = (__bf16*)alloc(ACT * 2);   // [B,H,S,dh]
  __bf16* Vtb = (__bf16*)alloc(ACT * 2);   // [B,H,dh,S]
  __bf16* Ab  = (__bf16*)alloc(ACT * 2);   // [B,S,D]

  cvt_bf16<<<(int)((ACT + 255) / 256), 256, 0, stream>>>(query, xbf, (int)ACT);
  cvt_bf16<<<(int)((WEL + 255) / 256), 256, 0, stream>>>(Wq, wqb, (int)WEL);
  cvt_bf16<<<(int)((WEL + 255) / 256), 256, 0, stream>>>(Wk, wkb, (int)WEL);
  cvt_bf16<<<(int)((WEL + 255) / 256), 256, 0, stream>>>(Wv, wvb, (int)WEL);
  cvt_bf16<<<(int)((WEL + 255) / 256), 256, 0, stream>>>(Wo, wob, (int)WEL);

  qkv_proj<<<dim3(BB * SS / 128, DDIM / 128, 3), 256, 0, stream>>>(
      xbf, wqb, wkb, wvb, bq, bk, bv, Qb, Kb, Vtb);

  attn<<<dim3(SS / 16, HH, BB), 256, 0, stream>>>(Qb, Kb, Vtb, mask, Ab);

  out_proj<<<dim3(BB * SS / 128, DDIM / 128), 256, 0, stream>>>(Ab, wob, bo, (float*)d_out);
}
